// LearnedTimeDiffusion_33732673143831
// MI455X (gfx1250) — matrix operand order repaired and checked by reading, not compiled, not run on hardware
//
#include <hip/hip_runtime.h>
#include <hip/hip_bf16.h>

typedef __bf16 bf16_t;
typedef __attribute__((ext_vector_type(16))) __bf16 v16bf;
typedef __attribute__((ext_vector_type(8)))  __bf16 v8bf;
typedef __attribute__((ext_vector_type(8)))  float  v8f;
typedef __attribute__((ext_vector_type(4)))  float  v4f;

#define BGRAPH 8
#define VG     20000
#define KEIG   128
#define DCH    128

// ---- fp32 -> bf16: native convert (single v_cvt, RNE) ----
__device__ __forceinline__ bf16_t f2bf(float f) {
  return static_cast<bf16_t>(f);
}

// Assemble a WMMA operand from two contiguous 16B LDS reads (ds_load_b128 pair).
__device__ __forceinline__ v16bf ld_op(const bf16_t* p, int hioff) {
  v8bf lo = *(const v8bf*)p;
  v8bf hi = *(const v8bf*)(p + hioff);
  v16bf r;
#pragma unroll
  for (int j = 0; j < 8; ++j) { r[j] = lo[j]; r[j + 8] = hi[j]; }
  return r;
}

// ---------------- kernel 1: zero the x_spec accumulator ----------------
__global__ void zero_kernel(float* __restrict__ p, int n) {
  int i = blockIdx.x * blockDim.x + threadIdx.x;
  if (i < n) p[i] = 0.0f;
}

// ---------------- kernel 2: x_spec[b] += E_b^T @ (M x)_b ---------------
// 125 chunks/graph, 5 k-steps of 32 vertices; double-buffered LDS so the
// next step's global loads overlap the WMMA burst. Wave mapping:
// 2 k-tiles x 4 d-tiles per wave -> 12 ds_b128 per 8 WMMAs.
#define P2_SPLIT  125
#define P2_STEPS  5
#define P2_STRIDE 40           // 32 + 8 pad (bf16); row pitch 80B (16B aligned)
#define P2_BUF    (128 * P2_STRIDE)

__global__ __launch_bounds__(256)
void to_basis_kernel(const float* __restrict__ x,
                     const float* __restrict__ mass,
                     const float* __restrict__ evecs,
                     float* __restrict__ xspec) {
  __shared__ __align__(16) bf16_t ldsE[2 * P2_BUF];   // E^T  [k][v], double buffered
  __shared__ __align__(16) bf16_t ldsX[2 * P2_BUF];   // (mx)^T [d][v]

  const int b     = blockIdx.x / P2_SPLIT;
  const int chunk = blockIdx.x % P2_SPLIT;
  const int t    = threadIdx.x;
  const int wave = t >> 5;
  const int lane = t & 31;
  const int m    = lane & 15;
  const int hi   = lane >> 4;
  const int kgrp = wave & 3;    // k-tiles {2*kgrp, 2*kgrp+1}
  const int dgrp = wave >> 2;   // d-tiles {4*dgrp .. 4*dgrp+3}

  // staging mapping: 8 threads per vertex row, 16 consecutive channels each
  const int srow = t >> 3;            // 0..31
  const int scol = (t & 7) * 16;      // 0..112
  const long vbase0 = (long)b * VG + (long)chunk * (P2_STEPS * 32);

  float eR[16], xR[16], mR;
  {   // prologue: prefetch step 0 into registers (coalesced b128 loads)
    const long v = vbase0 + srow;
    mR = mass[v];
    const v4f* ep = (const v4f*)(evecs + v * KEIG + scol);
    const v4f* xp = (const v4f*)(x + v * DCH + scol);
#pragma unroll
    for (int q = 0; q < 4; ++q) {
      v4f e4 = ep[q], x4 = xp[q];
#pragma unroll
      for (int j = 0; j < 4; ++j) { eR[q*4+j] = e4[j]; xR[q*4+j] = x4[j]; }
    }
  }

  v8f acc[8] = {};   // [k-tile local (2)] x [d-tile local (4)]

  for (int s = 0; s < P2_STEPS; ++s) {
    bf16_t* LE = ldsE + (s & 1) * P2_BUF;
    bf16_t* LX = ldsX + (s & 1) * P2_BUF;
    // write staged registers -> transposed bf16 LDS tiles
#pragma unroll
    for (int i = 0; i < 16; ++i) {
      LE[(scol + i) * P2_STRIDE + srow] = f2bf(eR[i]);
      LX[(scol + i) * P2_STRIDE + srow] = f2bf(mR * xR[i]);
    }
    __syncthreads();
    // prefetch next step while this step computes (uniform branch, EXEC stays full)
    if (s + 1 < P2_STEPS) {
      const long v = vbase0 + (s + 1) * 32 + srow;
      mR = mass[v];
      const v4f* ep = (const v4f*)(evecs + v * KEIG + scol);
      const v4f* xp = (const v4f*)(x + v * DCH + scol);
#pragma unroll
      for (int q = 0; q < 4; ++q) {
        v4f e4 = ep[q], x4 = xp[q];
#pragma unroll
        for (int j = 0; j < 4; ++j) { eR[q*4+j] = e4[j]; xR[q*4+j] = x4[j]; }
      }
    }
    // batch all operands into registers, then one WMMA burst
    v16bf A[2], B[4];
#pragma unroll
    for (int kl = 0; kl < 2; ++kl)
      A[kl] = ld_op(&LE[((2*kgrp + kl) * 16 + m) * P2_STRIDE] + hi * 8, 16);
#pragma unroll
    for (int dl = 0; dl < 4; ++dl)
      B[dl] = ld_op(&LX[((4*dgrp + dl) * 16 + m) * P2_STRIDE] + hi * 8, 16);
#pragma unroll
    for (int kl = 0; kl < 2; ++kl)
#pragma unroll
      for (int dl = 0; dl < 4; ++dl)
        acc[kl*4 + dl] = __builtin_amdgcn_wmma_f32_16x16x32_bf16(
            false, A[kl], false, B[dl], (short)0, acc[kl*4 + dl], false, false);
  }

  // accumulate partial (k,d) tiles into global x_spec (512KB, L2-resident)
#pragma unroll
  for (int kl = 0; kl < 2; ++kl)
#pragma unroll
    for (int dl = 0; dl < 4; ++dl)
#pragma unroll
      for (int r = 0; r < 8; ++r) {
        const int k = (2*kgrp + kl) * 16 + hi * 8 + r;
        const int d = (4*dgrp + dl) * 16 + (lane & 15);
        atomicAdd(&xspec[((long)b * KEIG + k) * DCH + d], acc[kl*4 + dl][r]);
      }
}

// ---------------- kernel 3: S = bf16( exp(-lambda*t) * x_spec ) --------
__global__ void scale_kernel(const float* __restrict__ xspec,
                             const float* __restrict__ evals,
                             const float* __restrict__ tim,
                             bf16_t* __restrict__ S) {
  int i = blockIdx.x * blockDim.x + threadIdx.x;
  if (i >= BGRAPH * KEIG * DCH) return;
  const int d = i & 127;
  const int k = (i >> 7) & 127;
  const int b = i >> 14;
  const float tt = fmaxf(tim[d], 1e-8f);
  const float c  = __expf(-evals[b * KEIG + k] * tt);
  S[i] = f2bf(c * xspec[i]);
}

// ---------------- kernel 4: out[b] = E_b @ S_b -------------------------
// 160-vertex tiles (125 chunks/graph). Wave mapping: 2 d-tiles x 5 m-tiles
// per wave -> 14 ds_b128 per 10 WMMAs, operands batched per k-step.
#define P4_CHUNKS 125
#define P4_MTILE  160
#define P4_STRIDE 136          // 128 + 8 pad (bf16); row pitch 272B (16B aligned)

__global__ __launch_bounds__(256)
void from_basis_kernel(const float* __restrict__ evecs,
                       const bf16_t* __restrict__ S,
                       float* __restrict__ out) {
  __shared__ __align__(16) bf16_t Sld[128 * P4_STRIDE];       // S^T : [d][k]
  __shared__ __align__(16) bf16_t El [P4_MTILE * P4_STRIDE];  // E   : [v][k]

  const int b     = blockIdx.x / P4_CHUNKS;
  const int chunk = blockIdx.x % P4_CHUNKS;
  const int t    = threadIdx.x;
  const int wave = t >> 5;
  const int lane = t & 31;
  const int m    = lane & 15;
  const int hi   = lane >> 4;
  const int dgrp = wave & 3;    // d-tiles {2*dgrp, 2*dgrp+1}
  const int mgrp = wave >> 2;   // m-tiles {5*mgrp .. 5*mgrp+4}

  const long vbase = (long)b * VG + (long)chunk * P4_MTILE;
  const bf16_t* Sb = S + (long)b * KEIG * DCH;

  for (int idx = t; idx < KEIG * DCH; idx += 256) {
    const int k = idx >> 7, d = idx & 127;
    Sld[d * P4_STRIDE + k] = Sb[idx];
  }
  for (int idx = t; idx < P4_MTILE * KEIG; idx += 256) {
    const int vl = idx >> 7, c = idx & 127;
    El[vl * P4_STRIDE + c] = f2bf(evecs[(vbase + vl) * KEIG + c]);
  }
  __syncthreads();

  v8f acc[10] = {};   // [m-tile local (5)] x [d-tile local (2)]

#pragma unroll
  for (int ks = 0; ks < 4; ++ks) {
    const int kb = ks * 32;
    v16bf Bt[2], At[5];
#pragma unroll
    for (int dl = 0; dl < 2; ++dl)
      Bt[dl] = ld_op(&Sld[((2*dgrp + dl) * 16 + m) * P4_STRIDE] + kb + hi * 8, 16);
#pragma unroll
    for (int ml = 0; ml < 5; ++ml)
      At[ml] = ld_op(&El[((5*mgrp + ml) * 16 + m) * P4_STRIDE] + kb + hi * 8, 16);
#pragma unroll
    for (int ml = 0; ml < 5; ++ml)
#pragma unroll
      for (int dl = 0; dl < 2; ++dl)
        acc[ml*2 + dl] = __builtin_amdgcn_wmma_f32_16x16x32_bf16(
            false, At[ml], false, Bt[dl], (short)0, acc[ml*2 + dl], false, false);
  }

#pragma unroll
  for (int ml = 0; ml < 5; ++ml)
#pragma unroll
    for (int dl = 0; dl < 2; ++dl)
#pragma unroll
      for (int r = 0; r < 8; ++r) {
        const long v = vbase + (5*mgrp + ml) * 16 + hi * 8 + r;
        out[v * DCH + (2*dgrp + dl) * 16 + (lane & 15)] = acc[ml*2 + dl][r];
      }
}

// ---------------- launcher --------------------------------------------
extern "C" void kernel_launch(void* const* d_in, const int* in_sizes, int n_in,
                              void* d_out, int out_size, void* d_ws, size_t ws_size,
                              hipStream_t stream) {
  const float* x     = (const float*)d_in[0];
  const float* mass  = (const float*)d_in[1];
  const float* evals = (const float*)d_in[2];
  const float* evecs = (const float*)d_in[3];
  const float* tim   = (const float*)d_in[4];
  // d_in[5] = batch (unused: equal-sized graphs), d_in[6] = bs (hardcoded)
  float* out = (float*)d_out;

  float*  xspec = (float*)d_ws;                                                    // 512KB fp32
  bf16_t* S     = (bf16_t*)((char*)d_ws + (size_t)BGRAPH*KEIG*DCH*sizeof(float));  // 256KB bf16

  const int NSPEC = BGRAPH * KEIG * DCH;
  zero_kernel<<<(NSPEC + 255) / 256, 256, 0, stream>>>(xspec, NSPEC);
  to_basis_kernel<<<BGRAPH * P2_SPLIT, 256, 0, stream>>>(x, mass, evecs, xspec);
  scale_kernel<<<(NSPEC + 255) / 256, 256, 0, stream>>>(xspec, evals, tim, S);
  from_basis_kernel<<<BGRAPH * P4_CHUNKS, 256, 0, stream>>>(evecs, S, out);
}
